// SimpleGCNTanh_48361331753433
// MI455X (gfx1250) — matrix operand order repaired and checked by reading, not compiled, not run on hardware
//
#include <hip/hip_runtime.h>
#include <hip/hip_bf16.h>
#include <math.h>

typedef __attribute__((ext_vector_type(16))) _Float16 v16h;
typedef __attribute__((ext_vector_type(8)))  float    v8f;

#define N_NODES 100000
#define N_EDGES 1600000

// ---------------------------------------------------------------- utilities
__global__ void zero_f32_kernel(float* __restrict__ p, int n) {
  int i = blockIdx.x * blockDim.x + threadIdx.x;
  if (i < n) p[i] = 0.0f;
}

__global__ void deg_accum_kernel(const int* __restrict__ dst, float* __restrict__ deg, int nE) {
  int i = blockIdx.x * blockDim.x + threadIdx.x;
  if (i < nE) atomicAdd(&deg[dst[i]], 1.0f);
}

__global__ void dinv_fin_kernel(float* __restrict__ d, int n) {
  int i = blockIdx.x * blockDim.x + threadIdx.x;
  if (i < n) d[i] = rsqrtf(d[i] + 1.0f);
}

__device__ __forceinline__ void pack4(v16h& dst, int base, float4 f) {
  dst[base + 0] = (_Float16)f.x;
  dst[base + 1] = (_Float16)f.y;
  dst[base + 2] = (_Float16)f.z;
  dst[base + 3] = (_Float16)f.w;
}

// ---------------------------------------------------------------- WMMA GEMM
// OUT[rows x NOUT] = H[rows x K] @ W[K x NOUT]  (+bias, tanh if TANH)
// block = 256 threads (8 waves), each block covers 128 rows; each wave owns a
// 16-row strip and loops over 16-col tiles with v_wmma_f32_16x16x32_f16.
// Weights staged in LDS pre-swizzled into WMMA B-fragment order so each lane
// reads its whole fragment as one contiguous v16h (2x ds_load_b128).
// NOTE: requires rows % 8 == 0 (true here: rows = 100000), so every 8-row
// output slab is either fully in bounds or fully out -> single store path.
template <int K, int NOUT, bool TANH>
__global__ __launch_bounds__(256) void gemm_wmma_kernel(
    const float* __restrict__ H, const float* __restrict__ W,
    const float* __restrict__ bias, float* __restrict__ OUT, int rows) {
  constexpr int KSTEPS = K / 32;
  constexpr int NTILES = NOUT / 16;
  // ldsW[frag][lane][elem], frag = nt*KSTEPS + kk
  __shared__ _Float16 ldsW[NTILES * KSTEPS * 32 * 16];

  const int tid = threadIdx.x;
  // Cooperative fill in B-fragment order:
  //   elem e of lane L in frag (nt,kk) holds W[k][n] with
  //   n = nt*16 + (L&15), k = kk*32 + (L>>4)*8 + (e&7) + (e>>3)*16
  for (int idx = tid; idx < K * NOUT; idx += 256) {
    const int e    = idx & 15;
    const int L    = (idx >> 4) & 31;
    const int frag = idx >> 9;
    const int kk   = frag % KSTEPS;
    const int nt   = frag / KSTEPS;
    const int n = nt * 16 + (L & 15);
    const int k = kk * 32 + (L >> 4) * 8 + (e & 7) + (e >> 3) * 16;
    ldsW[idx] = (_Float16)W[k * NOUT + n];
  }
  __syncthreads();

  const int wave = tid >> 5;
  const int lane = tid & 31;
  const int hlf  = lane >> 4;   // half-wave select
  const int l    = lane & 15;
  const int rowBase = blockIdx.x * 128 + wave * 16;
  const int m = rowBase + l;    // A-matrix row held by this lane

  // A fragments: lane l holds row M=l; per ISA table elements 0..7 are
  // K = kk*32 + hlf*8 + 0..7 and elements 8..15 are K = +16 of that.
  v16h a[KSTEPS];
#pragma unroll
  for (int kk = 0; kk < KSTEPS; ++kk) {
    float4 f0 = {0, 0, 0, 0}, f1 = {0, 0, 0, 0}, f2 = {0, 0, 0, 0}, f3 = {0, 0, 0, 0};
    if (m < rows) {
      const float* rp = H + (long long)m * K + kk * 32 + hlf * 8;
      f0 = *(const float4*)(rp + 0);
      f1 = *(const float4*)(rp + 4);
      f2 = *(const float4*)(rp + 16);
      f3 = *(const float4*)(rp + 20);
    }
    pack4(a[kk], 0, f0);
    pack4(a[kk], 4, f1);
    pack4(a[kk], 8, f2);
    pack4(a[kk], 12, f3);
  }

#pragma unroll
  for (int nt = 0; nt < NTILES; ++nt) {
    const int n = nt * 16 + l;  // D column held by this lane
    v8f c = {};
#pragma unroll
    for (int kk = 0; kk < KSTEPS; ++kk) {
      const v16h b =
          *(const v16h*)(ldsW + (((nt * KSTEPS + kk) * 32 + lane) << 4));
      c = __builtin_amdgcn_wmma_f32_16x16x32_f16(false, a[kk], false, b,
                                                 (short)0, c, false, false);
    }

    // D store: M = vgpr + 8*hlf, N = lane&15. rows%8==0 -> slab is fully in
    // or fully out of bounds; single clause of 8 stores w/ immediate offsets.
    float bval = 0.0f;
    if constexpr (TANH) bval = bias[n];
    const int mBase = rowBase + hlf * 8;
    if (mBase + 8 <= rows) {
      float* op = OUT + (long long)mBase * NOUT + n;
#pragma unroll
      for (int r = 0; r < 8; ++r) {
        float val = c[r];
        if constexpr (TANH) val = tanhf(val + bval);
        op[r * NOUT] = val;
      }
    }
  }
}

// ------------------------------------------------- edge scatter (L2 atomics)
// one wave per edge: lane carries 4 features; agg[dst] += hw[src]*coef
__global__ __launch_bounds__(256) void edge_agg_kernel(
    const float* __restrict__ hw, const float* __restrict__ dinv,
    const int* __restrict__ src, const int* __restrict__ dst,
    float* __restrict__ agg, int nE) {
  long long t = (long long)blockIdx.x * blockDim.x + threadIdx.x;
  int e = (int)(t >> 5);
  int lane = (int)(t & 31);
  if (e >= nE) return;
  const int s = src[e], d = dst[e];
  const float coef = dinv[s] * dinv[d];
  const float4 v = *(const float4*)(hw + (long long)s * 128 + lane * 4);
  float* ap = agg + (long long)d * 128 + lane * 4;
  atomicAdd(ap + 0, v.x * coef);
  atomicAdd(ap + 1, v.y * coef);
  atomicAdd(ap + 2, v.z * coef);
  atomicAdd(ap + 3, v.w * coef);
}

// h = tanh(agg + hw*dinv^2 + b)
__global__ void gcn_finalize_kernel(const float* __restrict__ hw,
                                    const float* __restrict__ agg,
                                    const float* __restrict__ dinv,
                                    const float* __restrict__ bias,
                                    float* __restrict__ hout, int total) {
  int i = blockIdx.x * blockDim.x + threadIdx.x;
  if (i >= total) return;
  const int node = i >> 7;
  const int f = i & 127;
  const float di = dinv[node];
  hout[i] = tanhf(agg[i] + hw[i] * di * di + bias[f]);
}

// ------------------------------------------------- edge classifier + concat
__global__ __launch_bounds__(256) void edge_cls_kernel(
    const float* __restrict__ h3, const int* __restrict__ src,
    const int* __restrict__ dst, const float* __restrict__ Wcls,
    const float* __restrict__ bcls, float* __restrict__ out,
    float* __restrict__ eout, int nE) {
  __shared__ float sW[64 * 6];
  __shared__ float sB[6];
  for (int i = threadIdx.x; i < 64 * 6; i += 256) sW[i] = Wcls[i];
  if (threadIdx.x < 6) sB[threadIdx.x] = bcls[threadIdx.x];
  __syncthreads();

  const int e = blockIdx.x * blockDim.x + threadIdx.x;
  if (e >= nE) return;
  const int s = src[e], d = dst[e];

  float ev[64];
  const float4* ps = (const float4*)(h3 + (long long)s * 32);
  const float4* pd = (const float4*)(h3 + (long long)d * 32);
#pragma unroll
  for (int q = 0; q < 8; ++q) {
    float4 t = ps[q];
    ev[4 * q + 0] = t.x; ev[4 * q + 1] = t.y; ev[4 * q + 2] = t.z; ev[4 * q + 3] = t.w;
  }
#pragma unroll
  for (int q = 0; q < 8; ++q) {
    float4 t = pd[q];
    ev[32 + 4 * q + 0] = t.x; ev[32 + 4 * q + 1] = t.y;
    ev[32 + 4 * q + 2] = t.z; ev[32 + 4 * q + 3] = t.w;
  }

  float acc[6];
#pragma unroll
  for (int c = 0; c < 6; ++c) acc[c] = sB[c];
#pragma unroll
  for (int k = 0; k < 64; ++k) {
#pragma unroll
    for (int c = 0; c < 6; ++c) acc[c] = fmaf(ev[k], sW[k * 6 + c], acc[c]);
  }
#pragma unroll
  for (int c = 0; c < 6; ++c) out[(long long)e * 6 + c] = acc[c];

  float4* pe = (float4*)(eout + (long long)e * 64);
#pragma unroll
  for (int q = 0; q < 16; ++q)
    pe[q] = make_float4(ev[4 * q], ev[4 * q + 1], ev[4 * q + 2], ev[4 * q + 3]);
}

// ---------------------------------------------------------------- launcher
extern "C" void kernel_launch(void* const* d_in, const int* in_sizes, int n_in,
                              void* d_out, int out_size, void* d_ws, size_t ws_size,
                              hipStream_t stream) {
  (void)in_sizes; (void)n_in; (void)out_size; (void)ws_size;

  const float* x    = (const float*)d_in[0];
  const int*   ei   = (const int*)d_in[1];      // [2, E]
  const float* W1   = (const float*)d_in[3];
  const float* b1   = (const float*)d_in[4];
  const float* Wc   = (const float*)d_in[5];    // [2,128,128]
  const float* bc   = (const float*)d_in[6];    // [2,128]
  const float* W2   = (const float*)d_in[7];
  const float* b2   = (const float*)d_in[8];
  const float* W3   = (const float*)d_in[9];
  const float* b3   = (const float*)d_in[10];
  const float* Wcls = (const float*)d_in[11];
  const float* bcls = (const float*)d_in[12];

  const int* srcI = ei;
  const int* dstI = ei + N_EDGES;

  // workspace layout (f32): dinv[N] | bufA[N*128] | bufB[N*128] | bufC[N*128]
  float* dinv = (float*)d_ws;
  float* bufA = dinv + N_NODES;
  float* bufB = bufA + (size_t)N_NODES * 128;
  float* bufC = bufB + (size_t)N_NODES * 128;

  const int NF = N_NODES * 128;
  const int gemmGrid = (N_NODES + 127) / 128;

  // degrees -> dinv = (deg+1)^-1/2
  zero_f32_kernel<<<(N_NODES + 255) / 256, 256, 0, stream>>>(dinv, N_NODES);
  deg_accum_kernel<<<(N_EDGES + 255) / 256, 256, 0, stream>>>(dstI, dinv, N_EDGES);
  dinv_fin_kernel<<<(N_NODES + 255) / 256, 256, 0, stream>>>(dinv, N_NODES);

  auto gcn_layer = [&](const float* hin, const float* Wl, const float* bl) {
    gemm_wmma_kernel<128, 128, false><<<gemmGrid, 256, 0, stream>>>(hin, Wl, nullptr, bufB, N_NODES);
    zero_f32_kernel<<<(NF + 255) / 256, 256, 0, stream>>>(bufC, NF);
    edge_agg_kernel<<<(N_EDGES * 32 + 255) / 256, 256, 0, stream>>>(bufB, dinv, srcI, dstI, bufC, N_EDGES);
    gcn_finalize_kernel<<<(NF + 255) / 256, 256, 0, stream>>>(bufB, bufC, dinv, bl, bufA, NF);
  };

  gcn_layer(x, W1, b1);
  gcn_layer(bufA, Wc, bc);
  gcn_layer(bufA, Wc + 128 * 128, bc + 128);

  gemm_wmma_kernel<128, 64, true><<<gemmGrid, 256, 0, stream>>>(bufA, W2, b2, bufB, N_NODES);
  gemm_wmma_kernel<64, 32, true><<<gemmGrid, 256, 0, stream>>>(bufB, W3, b3, bufC, N_NODES);

  float* outp = (float*)d_out;
  edge_cls_kernel<<<(N_EDGES + 255) / 256, 256, 0, stream>>>(
      bufC, srcI, dstI, Wcls, bcls, outp, outp + (size_t)N_EDGES * 6, N_EDGES);
}
